// IDDecoder_19937238188653
// MI455X (gfx1250) — matrix operand order repaired and checked by reading, not compiled, not run on hardware
//
#include <hip/hip_runtime.h>

// ---------------- model constants ----------------
#define B_   4
#define T_   16
#define N_   256
#define M_   128
#define G_   8
#define D_   512
#define H_   8
#define HD_  64
#define FFN_ 1024
#define FD_  256
#define NL_  6
#define FR_  (B_*T_)        // 64 frames
#define TOK_ (B_*T_*N_)     // 16384 tokens
#define SCALE_ 0.125f       // HD^-0.5
#define EPS_   1e-5f

typedef __bf16 bf16_t;
typedef __attribute__((ext_vector_type(16))) __bf16 v16bf;
typedef __attribute__((ext_vector_type(8)))  __bf16 v8bf;
typedef __attribute__((ext_vector_type(8)))  float  v8f;
typedef __attribute__((ext_vector_type(4)))  float  v4f;

// =====================================================================
// WMMA GEMM: C[m,n] = alpha * sum_k A[m,k]*B[n,k] + bias[n]  (optional ReLU)
// A: (M,K) row-major bf16, lda; B: (N,K) row-major bf16, ldb (K contiguous).
// Block = 256 threads = 8 waves (4x2); block tile 128x64; wave tile 32x32
// (2 A-frags x 2 B-frags -> 4 WMMA accumulators, each frag reused twice).
// All Mr multiples of 128, Nc multiples of 64, K multiples of 32 here.
// Batched via grid.z with two-level strides: z -> (z/zdiv, z%zdiv).
// =====================================================================
template<bool RELU>
__global__ __launch_bounds__(256)
void gemm_wmma(const bf16_t* __restrict__ A, const bf16_t* __restrict__ Bm,
               float* __restrict__ C, bf16_t* __restrict__ Cb,
               const float* __restrict__ bias,
               int Mr, int Nc, int K, int lda, int ldb, int ldc, float alpha,
               int zdiv,
               long long sA1, long long sA2, long long sB1, long long sB2,
               long long sC1, long long sC2)
{
    const int z  = blockIdx.z;
    const int z1 = z / zdiv, z2 = z - z1 * zdiv;
    A  += (long long)z1 * sA1 + (long long)z2 * sA2;
    Bm += (long long)z1 * sB1 + (long long)z2 * sB2;
    const long long cofs = (long long)z1 * sC1 + (long long)z2 * sC2;

    const int wave = threadIdx.x >> 5;
    const int lane = threadIdx.x & 31;
    const int r    = lane & 15;      // row/col within 16
    const int hi   = lane >> 4;      // half-select

    const int row0 = blockIdx.y * 128 + (wave >> 1) * 32;
    const int col0 = blockIdx.x * 64  + (wave & 1)  * 32;
    if (row0 >= Mr || col0 >= Nc) return;

    v8f acc00 = {}, acc01 = {}, acc10 = {}, acc11 = {};

    const bf16_t* aptr0 = A  + (size_t)(row0 +      r) * lda;
    const bf16_t* aptr1 = A  + (size_t)(row0 + 16 + r) * lda;
    const bf16_t* bptr0 = Bm + (size_t)(col0 +      r) * ldb;
    const bf16_t* bptr1 = Bm + (size_t)(col0 + 16 + r) * ldb;

    for (int k0 = 0; k0 < K; k0 += 32) {
        // branch-free clamped prefetch (always in-bounds)
        const int kpf = (k0 + 64 < K) ? (k0 + 64) : 0;
        __builtin_prefetch(aptr0 + kpf, 0, 1);   // global_prefetch_b8
        __builtin_prefetch(bptr0 + kpf, 0, 1);

        // A fragment (16x32): lane holds one row, K = k0 + hi*16 .. +15 contiguous
        union { v16bf v; v8bf h[2]; } a0, a1;
        a0.v = *(const v16bf*)(aptr0 + k0 + hi * 16);
        a1.v = *(const v16bf*)(aptr1 + k0 + hi * 16);
        // B fragment (32x16): lane holds one output col, K = k0 + hi*16 .. +15 contiguous
        const v16bf b0 = *(const v16bf*)(bptr0 + k0 + hi * 16);
        const v16bf b1 = *(const v16bf*)(bptr1 + k0 + hi * 16);

        acc00 = __builtin_amdgcn_wmma_f32_16x16x32_bf16(false, a0.v, false, b0, (short)0, acc00, false, false);
        acc01 = __builtin_amdgcn_wmma_f32_16x16x32_bf16(false, a0.v, false, b1, (short)0, acc01, false, false);
        acc10 = __builtin_amdgcn_wmma_f32_16x16x32_bf16(false, a1.v, false, b0, (short)0, acc10, false, false);
        acc11 = __builtin_amdgcn_wmma_f32_16x16x32_bf16(false, a1.v, false, b1, (short)0, acc11, false, false);
    }

    const float bi0 = bias ? bias[col0 +      r] : 0.0f;
    const float bi1 = bias ? bias[col0 + 16 + r] : 0.0f;
    float*  cpt = C  + cofs;
    bf16_t* cbt = Cb ? Cb + cofs : nullptr;
    #pragma unroll
    for (int i = 0; i < 8; ++i) {
        const int rrA = row0 + i + hi * 8;        // acc row map: vgpr i -> row i (+8 hi lanes)
        const int rrB = rrA + 16;
        float v00 = alpha * acc00[i] + bi0;
        float v01 = alpha * acc01[i] + bi1;
        float v10 = alpha * acc10[i] + bi0;
        float v11 = alpha * acc11[i] + bi1;
        if (RELU) {
            v00 = fmaxf(v00, 0.0f); v01 = fmaxf(v01, 0.0f);
            v10 = fmaxf(v10, 0.0f); v11 = fmaxf(v11, 0.0f);
        }
        cpt[(size_t)rrA * ldc + col0 +      r] = v00;
        cpt[(size_t)rrA * ldc + col0 + 16 + r] = v01;
        cpt[(size_t)rrB * ldc + col0 +      r] = v10;
        cpt[(size_t)rrB * ldc + col0 + 16 + r] = v11;
        if (cbt) {
            cbt[(size_t)rrA * ldc + col0 +      r] = (bf16_t)v00;
            cbt[(size_t)rrA * ldc + col0 + 16 + r] = (bf16_t)v01;
            cbt[(size_t)rrB * ldc + col0 +      r] = (bf16_t)v10;
            cbt[(size_t)rrB * ldc + col0 + 16 + r] = (bf16_t)v11;
        }
    }
}

// ---------------- LayerNorm (optional residual), emits f32 + optional bf16 ----
__global__ __launch_bounds__(256)
void ln_kernel(const float* __restrict__ x, const float* __restrict__ res,
               const float* __restrict__ g, const float* __restrict__ bt,
               float* __restrict__ out, bf16_t* __restrict__ outb, int Dd)
{
    __shared__ float s1[256], s2[256];
    const int row = blockIdx.x;
    const float* xr = x + (size_t)row * Dd;
    const float* rr = res ? res + (size_t)row * Dd : nullptr;
    float v[4];
    float a1 = 0.f, a2 = 0.f;
    int cnt = 0;
    for (int j = threadIdx.x; j < Dd; j += 256, ++cnt) {
        float t = xr[j] + (rr ? rr[j] : 0.f);
        v[cnt] = t; a1 += t; a2 += t * t;
    }
    s1[threadIdx.x] = a1; s2[threadIdx.x] = a2;
    __syncthreads();
    for (int o = 128; o > 0; o >>= 1) {
        if ((int)threadIdx.x < o) { s1[threadIdx.x] += s1[threadIdx.x + o];
                                    s2[threadIdx.x] += s2[threadIdx.x + o]; }
        __syncthreads();
    }
    const float mean = s1[0] / Dd;
    const float var  = s2[0] / Dd - mean * mean;
    const float rstd = rsqrtf(var + EPS_);
    cnt = 0;
    for (int j = threadIdx.x; j < Dd; j += 256, ++cnt) {
        float t = (v[cnt] - mean) * rstd * g[j] + bt[j];
        out[(size_t)row * Dd + j] = t;
        if (outb) outb[(size_t)row * Dd + j] = (bf16_t)t;
    }
}

// ---------------- row softmax: f32 in-place + bf16 copy -----------------------
// launched with blockDim.x == L (128 or 256)
__global__ void softmax_kernel(float* __restrict__ s, bf16_t* __restrict__ pb)
{
    __shared__ float sh[256];
    const int L = blockDim.x;
    const size_t row = blockIdx.x;
    const int t = threadIdx.x;
    float v = s[row * L + t];
    sh[t] = v; __syncthreads();
    for (int o = L >> 1; o > 0; o >>= 1) { if (t < o) sh[t] = fmaxf(sh[t], sh[t + o]); __syncthreads(); }
    const float mx = sh[0]; __syncthreads();
    float e = __expf(v - mx);
    sh[t] = e; __syncthreads();
    for (int o = L >> 1; o > 0; o >>= 1) { if (t < o) sh[t] += sh[t + o]; __syncthreads(); }
    const float inv = 1.0f / sh[0];
    float pr = e * inv;
    s [row * L + t] = pr;
    pb[row * L + t] = (bf16_t)pr;
}

// ---------------- mean over G groups ------------------------------------------
__global__ void group_mean_kernel(const float* __restrict__ in, float* __restrict__ out,
                                  bf16_t* __restrict__ outb, int n, int stride, int g)
{
    int i = blockIdx.x * blockDim.x + threadIdx.x;
    if (i >= n) return;
    float s = 0.f;
    for (int j = 0; j < g; ++j) s += in[(size_t)j * stride + i];
    s *= (1.0f / g);
    out[i] = s; outb[i] = (bf16_t)s;
}

// ---------------- attn mean over detections: (H,TOK,M) -> (FR,H,M) ------------
__global__ void attn_mean_kernel(const float* __restrict__ p, float* __restrict__ o)
{
    int idx = blockIdx.x * blockDim.x + threadIdx.x;
    if (idx >= FR_ * H_ * M_) return;
    int m  = idx % M_;
    int h  = (idx / M_) % H_;
    int bt = idx / (M_ * H_);
    const float* base = p + (size_t)h * TOK_ * M_ + (size_t)bt * N_ * M_ + m;
    float s = 0.f;
    for (int n = 0; n < N_; ++n) s += base[(size_t)n * M_];
    o[idx] = s * (1.0f / N_);
}

// ---------------- V transposes (so P·V GEMMs can use the clean T-mode path) ---
// SA: vt[((f*H+h)*HD + d) * N + m] = qkv_f32[(f*N+m)*1536 + 1024 + h*HD + d]
__global__ void sa_v_transpose_kernel(const float* __restrict__ qkv, bf16_t* __restrict__ vt)
{
    int idx = blockIdx.x * blockDim.x + threadIdx.x;
    if (idx >= TOK_ * D_) return;
    int m = idx % N_;
    int d = (idx / N_) % HD_;
    int h = (idx / (N_ * HD_)) % H_;
    int f =  idx / (N_ * HD_ * H_);
    vt[idx] = (bf16_t)qkv[((size_t)(f * N_ + m)) * 1536 + 1024 + h * HD_ + d];
}
// CA: vmt[dfull * M + m] = vme[m * D + dfull]
__global__ void ca_v_transpose_kernel(const float* __restrict__ vme, bf16_t* __restrict__ vmt)
{
    int idx = blockIdx.x * blockDim.x + threadIdx.x;
    if (idx >= D_ * M_) return;
    int m = idx % M_;
    int d = idx / M_;
    vmt[idx] = (bf16_t)vme[(size_t)m * D_ + d];
}

// ---------------- vectorized converts (n multiple of 8) -----------------------
__global__ void cast_bf16_kernel(const float* __restrict__ in, bf16_t* __restrict__ out, long long n8)
{
    long long i = (long long)blockIdx.x * blockDim.x + threadIdx.x;
    if (i >= n8) return;
    const v4f f0 = *(const v4f*)(in + i * 8);
    const v4f f1 = *(const v4f*)(in + i * 8 + 4);
    v8bf o;
    #pragma unroll
    for (int j = 0; j < 4; ++j) { o[j] = (bf16_t)f0[j]; o[j + 4] = (bf16_t)f1[j]; }
    *(v8bf*)(out + i * 8) = o;
}
__global__ void copy_cast_kernel(const float* __restrict__ in, float* __restrict__ of,
                                 bf16_t* __restrict__ ob, long long n8)
{
    long long i = (long long)blockIdx.x * blockDim.x + threadIdx.x;
    if (i >= n8) return;
    const v4f f0 = *(const v4f*)(in + i * 8);
    const v4f f1 = *(const v4f*)(in + i * 8 + 4);
    *(v4f*)(of + i * 8)     = f0;
    *(v4f*)(of + i * 8 + 4) = f1;
    v8bf o;
    #pragma unroll
    for (int j = 0; j < 4; ++j) { o[j] = (bf16_t)f0[j]; o[j + 4] = (bf16_t)f1[j]; }
    *(v8bf*)(ob + i * 8) = o;
}

// =====================================================================
// host side
// =====================================================================
static inline char* arena_take(char*& p, size_t bytes) {
    char* r = p; p += (bytes + 255) & ~(size_t)255; return r;
}

static void launch_gemm(hipStream_t st, const bf16_t* A, const bf16_t* B,
                        float* C, bf16_t* Cb, const float* bias,
                        int Mr, int Nc, int K, int lda, int ldb, int ldc, float alpha,
                        bool relu, int nz, int zdiv,
                        long long sA1, long long sA2, long long sB1, long long sB2,
                        long long sC1, long long sC2)
{
    dim3 g((Nc + 63) / 64, (Mr + 127) / 128, nz), b(256);
    if (relu) gemm_wmma<true ><<<g, b, 0, st>>>(A,B,C,Cb,bias,Mr,Nc,K,lda,ldb,ldc,alpha,zdiv,sA1,sA2,sB1,sB2,sC1,sC2);
    else      gemm_wmma<false><<<g, b, 0, st>>>(A,B,C,Cb,bias,Mr,Nc,K,lda,ldb,ldc,alpha,zdiv,sA1,sA2,sB1,sB2,sC1,sC2);
}

extern "C" void kernel_launch(void* const* d_in, const int* in_sizes, int n_in,
                              void* d_out, int out_size, void* d_ws, size_t ws_size,
                              hipStream_t stream)
{
    (void)in_sizes; (void)n_in; (void)out_size; (void)ws_size;
    const float* cur      = (const float*)d_in[0];
    const float* traj     = (const float*)d_in[1];
    const float* sa_in_w  = (const float*)d_in[2];
    const float* sa_in_b  = (const float*)d_in[3];
    const float* sa_out_w = (const float*)d_in[4];
    const float* sa_out_b = (const float*)d_in[5];
    const float* n1_g     = (const float*)d_in[6];
    const float* n1_b     = (const float*)d_in[7];
    const float* ca_qw    = (const float*)d_in[8];
    const float* ca_qb    = (const float*)d_in[9];
    const float* ca_kw    = (const float*)d_in[10];
    const float* ca_kb    = (const float*)d_in[11];
    const float* ca_vw    = (const float*)d_in[12];
    const float* ca_vb    = (const float*)d_in[13];
    const float* ca_ow    = (const float*)d_in[14];
    const float* ca_ob    = (const float*)d_in[15];
    const float* n2_g     = (const float*)d_in[16];
    const float* n2_b     = (const float*)d_in[17];
    const float* f_w1     = (const float*)d_in[18];
    const float* f_b1     = (const float*)d_in[19];
    const float* f_w2     = (const float*)d_in[20];
    const float* f_b2     = (const float*)d_in[21];
    const float* n3_g     = (const float*)d_in[22];
    const float* n3_b     = (const float*)d_in[23];
    const float* out_w    = (const float*)d_in[24];
    const float* out_b    = (const float*)d_in[25];
    const float* on_g     = (const float*)d_in[26];
    const float* on_b     = (const float*)d_in[27];

    float* out    = (float*)d_out;
    float* attn_o = out + (size_t)TOK_ * FD_;

    // ---- workspace carve ----
    char* p = (char*)d_ws;
    float*  X    = (float*) arena_take(p, (size_t)TOK_ * D_ * 4);
    bf16_t* Xb   = (bf16_t*)arena_take(p, (size_t)TOK_ * D_ * 2);
    float*  B1   = (float*) arena_take(p, (size_t)TOK_ * 1536 * 4);
    bf16_t* B1b  = (bf16_t*)arena_take(p, (size_t)TOK_ * 1536 * 2);
    float*  B2   = (float*) arena_take(p, (size_t)TOK_ * D_ * 4);
    bf16_t* B2b  = (bf16_t*)arena_take(p, (size_t)TOK_ * D_ * 2);
    float*  S    = (float*) arena_take(p, (size_t)FR_ * H_ * N_ * N_ * 4); // also CA scores
    bf16_t* Pb   = (bf16_t*)arena_take(p, (size_t)FR_ * H_ * N_ * N_ * 2);
    bf16_t* Vtb  = (bf16_t*)arena_take(p, (size_t)TOK_ * D_ * 2);          // SA V^T (bf16)
    bf16_t* Wb   = (bf16_t*)arena_take(p, (size_t)1536 * 512 * 2);
    bf16_t* Tb   = (bf16_t*)arena_take(p, (size_t)G_ * M_ * D_ * 2);
    float*  TP   = (float*) arena_take(p, (size_t)G_ * M_ * D_ * 4);
    float*  Kme  = (float*) arena_take(p, (size_t)M_ * D_ * 4);
    bf16_t* Kmeb = (bf16_t*)arena_take(p, (size_t)M_ * D_ * 2);
    float*  Vme  = (float*) arena_take(p, (size_t)M_ * D_ * 4);
    bf16_t* Vmeb = (bf16_t*)arena_take(p, (size_t)M_ * D_ * 2);
    bf16_t* Vmtb = (bf16_t*)arena_take(p, (size_t)D_ * M_ * 2);            // CA Vmean^T
    float*  OutP = (float*) arena_take(p, (size_t)TOK_ * FD_ * 4);

    auto castW = [&](const float* w, long long n) {
        long long n8 = n / 8;
        cast_bf16_kernel<<<(unsigned)((n8 + 255) / 256), 256, 0, stream>>>(w, Wb, n8);
    };

    // ---- init: X = current_tokens (f32 + bf16), Tb = traj bf16 ----
    {
        long long n8 = (long long)TOK_ * D_ / 8;
        copy_cast_kernel<<<(unsigned)((n8 + 255) / 256), 256, 0, stream>>>(cur, X, Xb, n8);
        long long nt8 = (long long)G_ * M_ * D_ / 8;
        cast_bf16_kernel<<<(unsigned)((nt8 + 255) / 256), 256, 0, stream>>>(traj, Tb, nt8);
    }

    for (int i = 0; i < NL_; ++i) {
        // ---------------- self-attention (layers 1..5) ----------------
        if (i >= 1) {
            const int j = i - 1;
            // qkv = x @ sa_in_w^T + b   -> B1 (f32, ld 1536) + B1b (bf16)
            castW(sa_in_w + (size_t)j * 1536 * 512, 1536LL * 512);
            launch_gemm(stream, Xb, Wb, B1, B1b, sa_in_b + (size_t)j * 1536,
                        TOK_, 1536, 512, 512, 512, 1536, 1.0f, false, 1, 1, 0,0,0,0,0,0);
            // V^T (per frame,head) from f32 qkv
            sa_v_transpose_kernel<<<(TOK_ * D_ + 255) / 256, 256, 0, stream>>>(B1, Vtb);
            // S = scale * Q K^T per (frame,head): batched z = FR*H
            launch_gemm(stream, B1b, B1b + 512, S, nullptr, nullptr,
                        N_, N_, HD_, 1536, 1536, N_, SCALE_, false,
                        FR_ * H_, H_,
                        (long long)N_ * 1536, 64, (long long)N_ * 1536, 64,
                        (long long)H_ * N_ * N_, (long long)N_ * N_);
            softmax_kernel<<<FR_ * H_ * N_, N_, 0, stream>>>(S, Pb);
            // O = P V: B = V^T (HD x N), K contiguous -> T-mode
            launch_gemm(stream, Pb, Vtb, B2, B2b, nullptr,
                        N_, HD_, N_, N_, N_, 512, 1.0f, false,
                        FR_ * H_, H_,
                        (long long)H_ * N_ * N_, (long long)N_ * N_,
                        (long long)H_ * HD_ * N_, (long long)HD_ * N_,
                        (long long)N_ * 512, 64);
            // proj + residual LN
            castW(sa_out_w + (size_t)j * 512 * 512, 512LL * 512);
            launch_gemm(stream, B2b, Wb, B1, nullptr, sa_out_b + (size_t)j * 512,
                        TOK_, 512, 512, 512, 512, 512, 1.0f, false, 1, 1, 0,0,0,0,0,0);
            ln_kernel<<<TOK_, 256, 0, stream>>>(X, B1, n1_g + (size_t)j * 512, n1_b + (size_t)j * 512,
                                                X, Xb, D_);
        }
        // ---------------- cross-attention ----------------
        castW(ca_qw + (size_t)i * 512 * 512, 512LL * 512);
        launch_gemm(stream, Xb, Wb, B1, B1b, ca_qb + (size_t)i * 512,
                    TOK_, 512, 512, 512, 512, 512, 1.0f, false, 1, 1, 0,0,0,0,0,0);
        castW(ca_kw + (size_t)i * 512 * 512, 512LL * 512);
        launch_gemm(stream, Tb, Wb, TP, nullptr, ca_kb + (size_t)i * 512,
                    G_ * M_, 512, 512, 512, 512, 512, 1.0f, false, 1, 1, 0,0,0,0,0,0);
        group_mean_kernel<<<(M_ * D_ + 255) / 256, 256, 0, stream>>>(TP, Kme, Kmeb, M_ * D_, M_ * D_, G_);
        castW(ca_vw + (size_t)i * 512 * 512, 512LL * 512);
        launch_gemm(stream, Tb, Wb, TP, nullptr, ca_vb + (size_t)i * 512,
                    G_ * M_, 512, 512, 512, 512, 512, 1.0f, false, 1, 1, 0,0,0,0,0,0);
        group_mean_kernel<<<(M_ * D_ + 255) / 256, 256, 0, stream>>>(TP, Vme, Vmeb, M_ * D_, M_ * D_, G_);
        ca_v_transpose_kernel<<<(D_ * M_ + 255) / 256, 256, 0, stream>>>(Vme, Vmtb);
        // S = scale * Q Kmean^T per head: (H, TOK, M)
        launch_gemm(stream, B1b, Kmeb, S, nullptr, nullptr,
                    TOK_, M_, HD_, 512, 512, M_, SCALE_, false,
                    H_, 1, 64, 0, 64, 0, (long long)TOK_ * M_, 0);
        softmax_kernel<<<H_ * TOK_, M_, 0, stream>>>(S, Pb);
        attn_mean_kernel<<<(FR_ * H_ * M_ + 255) / 256, 256, 0, stream>>>(
            S, attn_o + (size_t)i * FR_ * H_ * M_);
        // O = P Vmean: B = Vmean^T (HD x M) per head, T-mode
        launch_gemm(stream, Pb, Vmtb, B2, B2b, nullptr,
                    TOK_, HD_, M_, M_, M_, 512, 1.0f, false,
                    H_, 1, (long long)TOK_ * M_, 0, (long long)HD_ * M_, 0, 64, 0);
        // proj + residual LN
        castW(ca_ow + (size_t)i * 512 * 512, 512LL * 512);
        launch_gemm(stream, B2b, Wb, B1, nullptr, ca_ob + (size_t)i * 512,
                    TOK_, 512, 512, 512, 512, 512, 1.0f, false, 1, 1, 0,0,0,0,0,0);
        ln_kernel<<<TOK_, 256, 0, stream>>>(X, B1, n2_g + (size_t)i * 512, n2_b + (size_t)i * 512,
                                            X, Xb, D_);
        // ---------------- FFN ----------------
        castW(f_w1 + (size_t)i * FFN_ * 512, (long long)FFN_ * 512);
        launch_gemm(stream, Xb, Wb, B1, B1b, f_b1 + (size_t)i * FFN_,
                    TOK_, FFN_, 512, 512, 512, FFN_, 1.0f, true, 1, 1, 0,0,0,0,0,0);
        castW(f_w2 + (size_t)i * 512 * FFN_, 512LL * FFN_);
        launch_gemm(stream, B1b, Wb, B2, nullptr, f_b2 + (size_t)i * 512,
                    TOK_, 512, FFN_, FFN_, FFN_, 512, 1.0f, false, 1, 1, 0,0,0,0,0,0);
        ln_kernel<<<TOK_, 256, 0, stream>>>(X, B2, n3_g + (size_t)i * 512, n3_b + (size_t)i * 512,
                                            X, Xb, D_);
    }

    // ---------------- output head ----------------
    castW(out_w, (long long)FD_ * 512);
    launch_gemm(stream, Xb, Wb, OutP, nullptr, out_b,
                TOK_, FD_, 512, 512, 512, FD_, 1.0f, false, 1, 1, 0,0,0,0,0,0);
    ln_kernel<<<TOK_, 256, 0, stream>>>(OutP, nullptr, on_g, on_b, out, nullptr, FD_);
}
